// VQVAE_18717467476692
// MI455X (gfx1250) — matrix operand order repaired and checked by reading, not compiled, not run on hardware
//
#include <hip/hip_runtime.h>

// ---------------------------------------------------------------------------
// VQ-VAE forward for MI455X (gfx1250).
// Heavy contractions -> v_wmma_f32_16x16x32_bf16 (bf16 in, fp32 accum).
// Internal activation layout: NHWC, i.e. (rows = N*H*W, C=256) row-major fp32.
// ---------------------------------------------------------------------------

typedef __attribute__((ext_vector_type(16))) __bf16 v16bf;
typedef __attribute__((ext_vector_type(8)))  float  v8f;
typedef unsigned short u16_t;

#define BN_EPS 1e-5f

__device__ __forceinline__ u16_t f2bf(float f) {
  unsigned u = __float_as_uint(f);
  u += 0x7FFFu + ((u >> 16) & 1u);          // round-to-nearest-even
  return (u16_t)(u >> 16);
}

// ---------------------------------------------------------------------------
// Generic WMMA GEMM:  C[M,N] = A[M,K] * B[K,N] (+bias[n])
//   A  : bf16 row-major (M,K)
//   BT : bf16 row-major (N,K)  (i.e. B transposed; OIHW weights flatten to this)
//   C  : fp32, row `m` is remapped: orow = (m/(d0*d1))*s2 + ((m/d0)%d1)*s1
//                                        + (m%d0)*s0 + roff   (identity: d0 huge)
// Block: 256 threads = 8 wave32.  Tile 64(M) x 128(N) x 32(K).
// Wave (wm,wn) = (w>>2, w&3) owns a 32x32 sub-tile = 2x2 WMMA fragments.
// Requires: M%64==0, N%128==0, K%32==0 (true for every call below).
// ---------------------------------------------------------------------------
__global__ __launch_bounds__(256) void wmma_gemm(
    const u16_t* __restrict__ A, const u16_t* __restrict__ BT,
    float* __restrict__ C, const float* __restrict__ bias,
    int M, int N, int K, int d0, int d1, int s0, int s1, int s2, int roff)
{
  __shared__ uint4 As4[64 * 4];    // 64 rows * 32 bf16 (4 x uint4)  = 4 KiB
  __shared__ uint4 Bs4[128 * 4];   // 128 n-rows * 32 bf16           = 8 KiB

  const int tid  = threadIdx.x;
  const int lane = tid & 31, wave = tid >> 5;
  const int g    = lane >> 4, ln = lane & 15;     // half-wave group, lane-in-16
  const int wm   = wave >> 2, wn = wave & 3;
  const int m0   = blockIdx.x * 64;
  const int n0   = blockIdx.y * 128;
  const int arow = tid >> 2, aq = tid & 3;        // A tile: 256 uint4, 1/thread

  v8f acc[2][2] = {};

  for (int k0 = 0; k0 < K; k0 += 32) {
    // ---- global -> registers (prefetch) ----
    uint4 av  = ((const uint4*)(A  + (size_t)(m0 + arow) * K + k0))[aq];
    uint4 bv0 = ((const uint4*)(BT + (size_t)(n0 + (tid >> 2)) * K + k0))[tid & 3];
    int   u1  = tid + 256;
    uint4 bv1 = ((const uint4*)(BT + (size_t)(n0 + (u1  >> 2)) * K + k0))[u1 & 3];
    __syncthreads();
    As4[tid]       = av;
    Bs4[tid]       = bv0;
    Bs4[tid + 256] = bv1;
    __syncthreads();

    // ---- build fragments per ISA 7.12.2 lane layouts ----
    union Frag { uint4 u[2]; v16bf v; } afr[2], bfr[2];
#pragma unroll
    for (int tm = 0; tm < 2; ++tm) {
      int r = wm * 32 + tm * 16 + ln;            // A row (M index)
      afr[tm].u[0] = As4[r * 4 + g];             // K pairs {8g..8g+7}
      afr[tm].u[1] = As4[r * 4 + 2 + g];         // K pairs {16+8g..}
    }
#pragma unroll
    for (int tn = 0; tn < 2; ++tn) {
      int r = wn * 32 + tn * 16 + ln;            // B column (N index)
      bfr[tn].u[0] = Bs4[r * 4 + g];             // K {8g..8g+7}
      bfr[tn].u[1] = Bs4[r * 4 + 2 + g];         // K {16+8g..}
    }
#pragma unroll
    for (int tm = 0; tm < 2; ++tm)
#pragma unroll
      for (int tn = 0; tn < 2; ++tn)
        acc[tm][tn] = __builtin_amdgcn_wmma_f32_16x16x32_bf16(
            false, afr[tm].v, false, bfr[tn].v, (short)0, acc[tm][tn],
            false, false);
  }

  // ---- epilogue: C/D layout = VGPR r: M = r + 8*g, N = ln ----
  const int dd = d0 * d1;
#pragma unroll
  for (int tm = 0; tm < 2; ++tm)
#pragma unroll
    for (int tn = 0; tn < 2; ++tn) {
      int ng = n0 + wn * 32 + tn * 16 + ln;
      float bv = bias ? bias[ng] : 0.0f;
#pragma unroll
      for (int r = 0; r < 8; ++r) {
        int mg = m0 + wm * 32 + tm * 16 + g * 8 + r;
        int orow = (mg / dd) * s2 + ((mg / d0) % d1) * s1 + (mg % d0) * s0 + roff;
        C[(size_t)orow * N + ng] = acc[tm][tn][r] + bv;
      }
    }
}

// ---------------------------------------------------------------------------
// Packing / casting kernels
// ---------------------------------------------------------------------------

// OIHW fp32 -> BT[co][(ky*KW+kx)*Cin+ci] bf16
__global__ void pack_w_conv(const float* __restrict__ w, u16_t* __restrict__ bt,
                            int Cout, int Cin, int KH, int KW) {
  int K = KH * KW * Cin;
  long total = (long)Cout * K;
  long idx = (long)blockIdx.x * blockDim.x + threadIdx.x;
  if (idx >= total) return;
  int k = (int)(idx % K); int co = (int)(idx / K);
  int ci = k % Cin; int t = k / Cin; int kx = t % KW; int ky = t / KW;
  bt[idx] = f2bf(w[(((size_t)co * Cin + ci) * KH + ky) * KW + kx]);
}

// torch ConvT weight (Cin,Cout,4,4) -> per-parity-class BT[co][(ty*2+tx)*Cin+ci]
__global__ void pack_w_convt(const float* __restrict__ w, u16_t* __restrict__ bt,
                             int Cin, int Cout, int ky0, int ky1, int kx0, int kx1) {
  int K = 4 * Cin;
  long total = (long)Cout * K;
  long idx = (long)blockIdx.x * blockDim.x + threadIdx.x;
  if (idx >= total) return;
  int k = (int)(idx % K); int co = (int)(idx / K);
  int ci = k % Cin; int tap = k / Cin; int tx = tap & 1, ty = tap >> 1;
  int ky = ty ? ky1 : ky0, kx = tx ? kx1 : kx0;
  bt[idx] = f2bf(w[(((size_t)ci * Cout + co) * 4 + ky) * 4 + kx]);
}

__global__ void cast_bf16(const float* __restrict__ s, u16_t* __restrict__ d,
                          long n, int relu) {
  long i = (long)blockIdx.x * blockDim.x + threadIdx.x;
  if (i >= n) return;
  float v = s[i];
  if (relu) v = fmaxf(v, 0.0f);
  d[i] = f2bf(v);
}

// Implicit-GEMM A pack (im2col) from NHWC fp32; k = (ky*KW+kx)*C + ci.
__global__ void im2col_conv(const float* __restrict__ x, u16_t* __restrict__ A,
                            int Nb, int C, int H, int W, int OH, int OW,
                            int KH, int KW, int stride, int pad, int relu) {
  int K = KH * KW * C;
  long total = (long)Nb * OH * OW * K;
  long idx = (long)blockIdx.x * blockDim.x + threadIdx.x;
  if (idx >= total) return;
  int k = (int)(idx % K); long m = idx / K;
  int ci = k % C; int t = k / C; int kx = t % KW; int ky = t / KW;
  int ox = (int)(m % OW); long t2 = m / OW;
  int oy = (int)(t2 % OH); int n = (int)(t2 / OH);
  int iy = oy * stride - pad + ky, ix = ox * stride - pad + kx;
  float v = 0.0f;
  if ((unsigned)iy < (unsigned)H && (unsigned)ix < (unsigned)W)
    v = x[((size_t)(n * H + iy) * W + ix) * C + ci];
  if (relu) v = fmaxf(v, 0.0f);
  A[idx] = f2bf(v);
}

// ConvTranspose parity-class A pack: 2x2 taps, offsets dy/dx per tap.
__global__ void im2col_convt(const float* __restrict__ x, u16_t* __restrict__ A,
                             int Nb, int C, int S,
                             int dy0, int dy1, int dx0, int dx1, int relu) {
  long total = (long)Nb * S * S * 4 * C;
  long idx = (long)blockIdx.x * blockDim.x + threadIdx.x;
  if (idx >= total) return;
  int ci = (int)(idx % C); long t = idx / C;
  int tap = (int)(t & 3); long m = t >> 2;
  int tx = tap & 1, ty = tap >> 1;
  int xq = (int)(m % S); long t2 = m / S;
  int yq = (int)(t2 % S); int n = (int)(t2 / S);
  int iy = yq + (ty ? dy1 : dy0);
  int ix = xq + (tx ? dx1 : dx0);
  float v = 0.0f;
  if ((unsigned)iy < (unsigned)S && (unsigned)ix < (unsigned)S)
    v = x[((size_t)(n * S + iy) * S + ix) * C + ci];
  if (relu) v = fmaxf(v, 0.0f);
  A[idx] = f2bf(v);
}

// ---------------------------------------------------------------------------
// BatchNorm (batch statistics) on (M,256) NHWC fp32
// ---------------------------------------------------------------------------
__global__ void zero_f32(float* p, int n) {
  int i = blockIdx.x * blockDim.x + threadIdx.x;
  if (i < n) p[i] = 0.0f;
}

__global__ __launch_bounds__(256) void bn_stats(const float* __restrict__ x,
                                                float* __restrict__ sums, int M) {
  int c = threadIdx.x;                                  // channel (C==256)
  int chunk = (M + gridDim.x - 1) / gridDim.x;
  int m0 = blockIdx.x * chunk;
  int m1 = m0 + chunk; if (m1 > M) m1 = M;
  float s = 0.0f, s2 = 0.0f;
  for (int m = m0; m < m1; ++m) {
    float v = x[(size_t)m * 256 + c];
    s += v; s2 += v * v;
  }
  atomicAdd(&sums[c], s);
  atomicAdd(&sums[256 + c], s2);
}

__global__ void bn_apply(const float* __restrict__ h, float* __restrict__ dst,
                         const float* __restrict__ res, const float* __restrict__ sums,
                         const float* __restrict__ gamma, const float* __restrict__ beta,
                         int M, int relu) {
  size_t idx = (size_t)blockIdx.x * blockDim.x + threadIdx.x;
  if (idx >= (size_t)M * 256) return;
  int c = (int)(idx & 255);
  float invM = 1.0f / (float)M;
  float mean = sums[c] * invM;
  float var  = sums[256 + c] * invM - mean * mean;
  float y = (h[idx] - mean) * rsqrtf(var + BN_EPS) * gamma[c] + beta[c];
  if (relu) y = fmaxf(y, 0.0f);
  if (res)  y += res[idx];
  dst[idx] = y;
}

// ---------------------------------------------------------------------------
// Encoder conv1: 3->256, k4 s2 p1, direct (K=48, memory-bound). Out NHWC.
// Block = 256 threads (one per Cout); each block does 8 consecutive ox pixels.
// ---------------------------------------------------------------------------
__global__ __launch_bounds__(256) void conv1_k4(const float* __restrict__ img,
                                                const float* __restrict__ w,
                                                const float* __restrict__ bias,
                                                float* __restrict__ out, int Nb) {
  __shared__ float patch[3][4][18];
  int gid = blockIdx.x;
  int ox0 = (gid & 7) * 8;
  int oy  = (gid >> 3) & 63;
  int n   = gid >> 9;
  int tid = threadIdx.x;
  int ixb = 2 * ox0 - 1;
  if (tid < 216) {
    int ci = tid / 72; int r = tid % 72; int ky = r / 18; int lx = r % 18;
    int iy = 2 * oy - 1 + ky, ix = ixb + lx;
    float v = 0.0f;
    if ((unsigned)iy < 128u && (unsigned)ix < 128u)
      v = img[((size_t)(n * 3 + ci) * 128 + iy) * 128 + ix];
    patch[ci][ky][lx] = v;
  }
  __syncthreads();
  int co = tid;
  float acc[8];
#pragma unroll
  for (int p = 0; p < 8; ++p) acc[p] = bias[co];
  for (int ci = 0; ci < 3; ++ci)
#pragma unroll
    for (int ky = 0; ky < 4; ++ky)
#pragma unroll
      for (int kx = 0; kx < 4; ++kx) {
        float wv = w[(((size_t)co * 3 + ci) * 4 + ky) * 4 + kx];
#pragma unroll
        for (int p = 0; p < 8; ++p)
          acc[p] += wv * patch[ci][ky][2 * p + kx];
      }
  size_t row = (size_t)(n * 64 + oy) * 64 + ox0;
#pragma unroll
  for (int p = 0; p < 8; ++p) out[(row + p) * 256 + co] = acc[p];
}

// ---------------------------------------------------------------------------
// VQ: codebook norms, wave-per-row argmin of ||e||^2 - 2*dot, gather
// ---------------------------------------------------------------------------
__global__ void cnorm_k(const float* __restrict__ cb, float* __restrict__ cn,
                        int Kc, int C) {
  int n = blockIdx.x * blockDim.x + threadIdx.x;
  if (n >= Kc) return;
  float s = 0.0f;
  for (int c = 0; c < C; ++c) { float v = cb[(size_t)n * C + c]; s += v * v; }
  cn[n] = s;
}

__global__ __launch_bounds__(256) void vq_argmin(const float* __restrict__ dots,
                                                 const float* __restrict__ cnorm,
                                                 int* __restrict__ idx, int Kc) {
  int row  = blockIdx.x * 8 + (threadIdx.x >> 5);
  int lane = threadIdx.x & 31;
  float best = 3.4e38f; int bi = 0;
  for (int n = lane; n < Kc; n += 32) {
    float v = cnorm[n] - 2.0f * dots[(size_t)row * Kc + n];
    if (v < best) { best = v; bi = n; }
  }
#pragma unroll
  for (int off = 16; off > 0; off >>= 1) {
    float ov = __shfl_down(best, off, 32);
    int   oi = __shfl_down(bi,   off, 32);
    if (ov < best || (ov == best && oi < bi)) { best = ov; bi = oi; }
  }
  if (lane == 0) idx[row] = bi;
}

__global__ void vq_gather(const float* __restrict__ cb, const int* __restrict__ idx,
                          float* __restrict__ q, long n) {
  long i = (long)blockIdx.x * blockDim.x + threadIdx.x;
  if (i >= n) return;
  long m = i >> 8; int c = (int)(i & 255);
  q[i] = cb[(size_t)idx[m] * 256 + c];
}

// ---------------------------------------------------------------------------
// Decoder convT2: 256->3, k4 s2 p1, direct + tanh, output NCHW to d_out.
// Weights (256,3,4,4) staged in LDS (48 KiB).
// ---------------------------------------------------------------------------
__global__ __launch_bounds__(256) void convt2_tanh(const float* __restrict__ xup,
                                                   const float* __restrict__ w,
                                                   const float* __restrict__ bias,
                                                   float* __restrict__ out, int Nb) {
  __shared__ float wl[256 * 3 * 16];
  for (int i = threadIdx.x; i < 12288; i += 256) wl[i] = w[i];
  __syncthreads();
  int gidx = blockIdx.x * 256 + threadIdx.x;     // over Nb*128*128
  int ox = gidx & 127;
  int oy = (gidx >> 7) & 127;
  int n  = gidx >> 14;
  int yq = oy >> 1, xq = ox >> 1;
  int kys[2], dys[2], kxs[2], dxs[2];
  if ((oy & 1) == 0) { kys[0] = 1; dys[0] = 0; kys[1] = 3; dys[1] = -1; }
  else               { kys[0] = 0; dys[0] = 1; kys[1] = 2; dys[1] = 0;  }
  if ((ox & 1) == 0) { kxs[0] = 1; dxs[0] = 0; kxs[1] = 3; dxs[1] = -1; }
  else               { kxs[0] = 0; dxs[0] = 1; kxs[1] = 2; dxs[1] = 0;  }
  float a0 = bias[0], a1 = bias[1], a2 = bias[2];
#pragma unroll
  for (int ty = 0; ty < 2; ++ty) {
    int iy = yq + dys[ty];
    if ((unsigned)iy >= 64u) continue;
#pragma unroll
    for (int tx = 0; tx < 2; ++tx) {
      int ix = xq + dxs[tx];
      if ((unsigned)ix >= 64u) continue;
      const float* xrow = xup + ((size_t)(n * 64 + iy) * 64 + ix) * 256;
      int wofs = kys[ty] * 4 + kxs[tx];
      for (int ci = 0; ci < 256; ++ci) {
        float xv = xrow[ci];
        const float* wp = &wl[ci * 48 + wofs];
        a0 += xv * wp[0];
        a1 += xv * wp[16];
        a2 += xv * wp[32];
      }
    }
  }
  size_t base = ((size_t)n * 3) * 16384 + (size_t)oy * 128 + ox;
  out[base]           = tanhf(a0);
  out[base + 16384]   = tanhf(a1);
  out[base + 32768]   = tanhf(a2);
}

// ---------------------------------------------------------------------------
// Host orchestration
// ---------------------------------------------------------------------------
struct RBp { const float *b1, *b2, *bb1, *bb2, *g1, *g2, *w1, *w2; };

static inline unsigned gblk(long n) { return (unsigned)((n + 255) / 256); }

extern "C" void kernel_launch(void* const* d_in, const int* in_sizes, int n_in,
                              void* d_out, int out_size, void* d_ws, size_t ws_size,
                              hipStream_t stream) {
  (void)in_sizes; (void)n_in; (void)out_size;
  // JAX pytree flatten order (dict keys alphabetical):
  // 0 ori_image, 1 codebook, 2 d_b1, 3 d_b2, 4 d_bb1, 5 d_g1,
  // 6..13 d_rb1{b1,b2,bb1,bb2,g1,g2,w1,w2}, 14..21 d_rb2{...},
  // 22 d_w1, 23 d_w2, 24 e_b1, 25 e_b2, 26 e_bb1, 27 e_g1,
  // 28..35 e_rb1{...}, 36..43 e_rb2{...}, 44 e_w1, 45 e_w2
  const float* ori      = (const float*)d_in[0];
  const float* codebook = (const float*)d_in[1];
  const float* d_b1  = (const float*)d_in[2];
  const float* d_b2  = (const float*)d_in[3];
  const float* d_bb1 = (const float*)d_in[4];
  const float* d_g1  = (const float*)d_in[5];
  auto getRB = [&](int i0) {
    RBp r;
    r.b1  = (const float*)d_in[i0 + 0]; r.b2  = (const float*)d_in[i0 + 1];
    r.bb1 = (const float*)d_in[i0 + 2]; r.bb2 = (const float*)d_in[i0 + 3];
    r.g1  = (const float*)d_in[i0 + 4]; r.g2  = (const float*)d_in[i0 + 5];
    r.w1  = (const float*)d_in[i0 + 6]; r.w2  = (const float*)d_in[i0 + 7];
    return r;
  };
  RBp drb1 = getRB(6), drb2 = getRB(14), erb1 = getRB(28), erb2 = getRB(36);
  const float* d_w1  = (const float*)d_in[22];
  const float* d_w2  = (const float*)d_in[23];
  const float* e_b1  = (const float*)d_in[24];
  const float* e_b2  = (const float*)d_in[25];
  const float* e_bb1 = (const float*)d_in[26];
  const float* e_g1  = (const float*)d_in[27];
  const float* e_w1  = (const float*)d_in[44];
  const float* e_w2  = (const float*)d_in[45];

  const size_t MB = 1ull << 20;
  char* ws = (char*)d_ws;
  u16_t* WT   = (u16_t*)(ws);                 //   4 MiB  bf16 BT weights
  u16_t* IM   = (u16_t*)(ws + 4 * MB);        // 128 MiB  bf16 im2col A
  float* BIG  = (float*)(ws + 132 * MB);      //  64 MiB  (65536,256) fp32
  float* S0   = (float*)(ws + 196 * MB);      //  16 MiB  (16384,256)
  float* S1   = (float*)(ws + 212 * MB);
  float* S2   = (float*)(ws + 228 * MB);
  float* DOTS = (float*)(ws + 244 * MB);      //  32 MiB  (16384,512)
  float* STAT = (float*)(ws + 276 * MB);      // sums[512] + cnorm[512]
  int*   IDX  = (int*)(ws + 276 * MB + 4096); // 16384 ints
  if (ws_size < 277 * MB) return;

  const int ID0 = 1 << 30;   // identity row mapping

  auto bn = [&](const float* h, float* dst, const float* res,
                const float* gm, const float* bt, int M, int relu) {
    zero_f32<<<2, 256, 0, stream>>>(STAT, 512);
    bn_stats<<<64, 256, 0, stream>>>(h, STAT, M);
    bn_apply<<<gblk((long)M * 256), 256, 0, stream>>>(h, dst, res, STAT, gm, bt, M, relu);
  };
  auto gemm = [&](const u16_t* A, const u16_t* BT_, float* Cc, const float* bias,
                  int M, int N, int K, int d0, int d1, int s0, int s1, int s2,
                  int roff) {
    dim3 grd(M / 64, N / 128);
    wmma_gemm<<<grd, 256, 0, stream>>>(A, BT_, Cc, bias, M, N, K, d0, d1, s0, s1, s2, roff);
  };
  auto resblock = [&](const RBp& rb, float* S, int M, int HW) {
    pack_w_conv<<<gblk(256L * 2304), 256, 0, stream>>>(rb.w1, WT, 256, 256, 3, 3);
    im2col_conv<<<gblk((long)M * 2304), 256, 0, stream>>>(S, IM, 16, 256, HW, HW, HW, HW,
                                                          3, 3, 1, 1, /*relu*/1);
    gemm(IM, WT, S1, rb.b1, M, 256, 2304, ID0, 1, 1, 0, 0, 0);
    bn(S1, S1, nullptr, rb.g1, rb.bb1, M, /*relu*/1);
    cast_bf16<<<gblk((long)M * 256), 256, 0, stream>>>(S1, IM, (long)M * 256, 0);
    pack_w_conv<<<gblk(256L * 256), 256, 0, stream>>>(rb.w2, WT, 256, 256, 1, 1);
    gemm(IM, WT, S2, rb.b2, M, 256, 256, ID0, 1, 1, 0, 0, 0);
    bn(S2, S, S, rb.g2, rb.bb2, M, /*relu*/0);   // S = S + BN(h)
  };

  // ---------------- encoder ----------------
  conv1_k4<<<16 * 64 * 8, 256, 0, stream>>>(ori, e_w1, e_b1, BIG, 16);
  bn(BIG, BIG, nullptr, e_g1, e_bb1, 65536, /*relu*/1);

  pack_w_conv<<<gblk(256L * 4096), 256, 0, stream>>>(e_w2, WT, 256, 256, 4, 4);
  im2col_conv<<<gblk(16384L * 4096), 256, 0, stream>>>(BIG, IM, 16, 256, 64, 64, 32, 32,
                                                       4, 4, 2, 1, 0);
  gemm(IM, WT, S0, e_b2, 16384, 256, 4096, ID0, 1, 1, 0, 0, 0);

  resblock(erb1, S0, 16384, 32);
  resblock(erb2, S0, 16384, 32);

  // ---------------- vector quantizer ----------------
  cast_bf16<<<gblk(512L * 256), 256, 0, stream>>>(codebook, WT, 512L * 256, 0);
  cast_bf16<<<gblk(16384L * 256), 256, 0, stream>>>(S0, IM, 16384L * 256, 0);
  gemm(IM, WT, DOTS, nullptr, 16384, 512, 256, ID0, 1, 1, 0, 0, 0);
  cnorm_k<<<2, 256, 0, stream>>>(codebook, STAT + 512, 512, 256);
  vq_argmin<<<16384 / 8, 256, 0, stream>>>(DOTS, STAT + 512, IDX, 512);
  vq_gather<<<gblk(16384L * 256), 256, 0, stream>>>(codebook, IDX, S0, 16384L * 256);

  // ---------------- decoder ----------------
  resblock(drb1, S0, 16384, 32);
  resblock(drb2, S0, 16384, 32);

  // convT1 (k4 s2 p1, 256->256) = 4 parity-class GEMMs scattered into BIG(64x64)
  for (int py = 0; py < 2; ++py)
    for (int px = 0; px < 2; ++px) {
      int ky0 = py ? 0 : 1, ky1 = py ? 2 : 3, dy0 = py ? 1 : 0, dy1 = py ? 0 : -1;
      int kx0 = px ? 0 : 1, kx1 = px ? 2 : 3, dx0 = px ? 1 : 0, dx1 = px ? 0 : -1;
      pack_w_convt<<<gblk(256L * 1024), 256, 0, stream>>>(d_w1, WT, 256, 256,
                                                          ky0, ky1, kx0, kx1);
      im2col_convt<<<gblk(16384L * 1024), 256, 0, stream>>>(S0, IM, 16, 256, 32,
                                                            dy0, dy1, dx0, dx1, /*relu*/1);
      // m=(n,y,x) -> orow = n*4096 + y*128 + x*2 + (py*64+px)
      gemm(IM, WT, BIG, d_b1, 16384, 256, 1024, 32, 32, 2, 128, 4096, py * 64 + px);
    }
  bn(BIG, BIG, nullptr, d_g1, d_bb1, 65536, /*relu*/1);

  convt2_tanh<<<16 * 128 * 128 / 256, 256, 0, stream>>>(BIG, d_w2, d_b2,
                                                        (float*)d_out, 16);
}
